// Attention_55740085567875
// MI455X (gfx1250) — compile-verified, matrix operand order checked by reading, not run on hardware
//
#include <hip/hip_runtime.h>
#include <hip/hip_bf16.h>

// ---------------------------------------------------------------------------
// Attention block for MI455X (gfx1250, wave32, WMMA 16x16x32 f16 -> f32 acc)
// B=4, N=1024, C=1024, H=16, D=64.
// Outputs: [B,N,C] f32 proj output, then [B,H,N,N] f32 softmax weights.
// ---------------------------------------------------------------------------

typedef __attribute__((ext_vector_type(16))) _Float16 v16h;
typedef __attribute__((ext_vector_type(8)))  _Float16 v8h;
typedef __attribute__((ext_vector_type(8)))  float    v8f;

#define B_  4
#define N_  1024
#define C_  1024
#define H_  16
#define D_  64
#define LD3 3072   // 3*C

// Load a 16-lane-row x 32(K) f16 fragment in the CDNA5 WMMA per-lane layout:
// lane l (0..15): M=l, halves cover K = k0+{0..7} and k0+{16..23};
// lane l+16:      M=l, halves cover K = k0+{8..15} and k0+{24..31}.
// Works for the A operand of a row-major (K-contiguous) matrix, and for the
// B operand whenever the B *columns* are K-contiguous in memory (i.e. x@W^T
// with W stored row-major, or a transposed-staged V tile in LDS).
__device__ __forceinline__ v16h load_frag_K(const _Float16* __restrict__ base,
                                            int ld, int row, int k0, int lane) {
  const int m    = lane & 15;
  const int koff = (lane >> 4) << 3;
  const _Float16* p = base + (size_t)(row + m) * ld + k0 + koff;
  v8h lo = *reinterpret_cast<const v8h*>(p);
  v8h hi = *reinterpret_cast<const v8h*>(p + 16);
  v16h r;
#pragma unroll
  for (int i = 0; i < 8; ++i) { r[i] = lo[i]; r[8 + i] = hi[i]; }
  return r;
}

// ---------------------------------------------------------------------------
// f32 -> f16 conversion
// ---------------------------------------------------------------------------
__global__ void f32_to_f16_kernel(const float* __restrict__ in,
                                  _Float16* __restrict__ out, int n) {
  int i = blockIdx.x * blockDim.x + threadIdx.x;
  if (i < n) out[i] = (_Float16)in[i];
}

// ---------------------------------------------------------------------------
// Generic WMMA GEMM: Cout[m,j] = sum_k A[m,k] * Bw[j,k] + bias[j]
// Each wave computes a 32(M) x 64(N) register tile (2x4 WMMA tiles,
// 8 v8f accumulators): per K-chunk 6 fragment loads feed 8 WMMAs.
// Block = 256 threads (8 waves) spanning 32x512 of C per (blockIdx.x row).
// grid.x * 8 = N/64 column blocks, grid.y = M/32 row blocks.
// ---------------------------------------------------------------------------
template <bool OUT_F16>
__global__ void wmma_gemm_kernel(const _Float16* __restrict__ A, int lda,
                                 const _Float16* __restrict__ Bw, int ldb,
                                 const float* __restrict__ bias,
                                 void* __restrict__ Cout, int ldc, int K) {
  const int lane = threadIdx.x & 31;
  const int wave = threadIdx.x >> 5;
  const int colb = (blockIdx.x * 8 + wave) * 64;  // 64-wide column block
  const int rowb = blockIdx.y * 32;               // 32-tall row block

  v8f acc[2][4] = {};
  for (int k0 = 0; k0 < K; k0 += 32) {
    // prefetch next K-chunk into L2/L0 (global_prefetch_b8)
    if (k0 + 128 < K) {
      __builtin_prefetch(A  + (size_t)(rowb + (lane & 15)) * lda + k0 + 128, 0, 1);
      __builtin_prefetch(Bw + (size_t)(colb + (lane & 15)) * ldb + k0 + 128, 0, 1);
    }
    v16h a0 = load_frag_K(A, lda, rowb,      k0, lane);
    v16h a1 = load_frag_K(A, lda, rowb + 16, k0, lane);
#pragma unroll
    for (int j = 0; j < 4; ++j) {
      v16h b = load_frag_K(Bw, ldb, colb + j * 16, k0, lane);
      acc[0][j] = __builtin_amdgcn_wmma_f32_16x16x32_f16(false, a0, false, b,
                                                         (short)0, acc[0][j], false, false);
      acc[1][j] = __builtin_amdgcn_wmma_f32_16x16x32_f16(false, a1, false, b,
                                                         (short)0, acc[1][j], false, false);
    }
  }
  // C/D layout: lane l: N = l&15; VGPR r: M = r + 8*(l>>4)
  const int mb = ((lane >> 4) << 3);
#pragma unroll
  for (int j = 0; j < 4; ++j) {
    const int n = colb + j * 16 + (lane & 15);
    const float bv = bias[n];
#pragma unroll
    for (int mi = 0; mi < 2; ++mi) {
      const int mrow = rowb + mi * 16 + mb;
#pragma unroll
      for (int r = 0; r < 8; ++r) {
        float v = acc[mi][j][r] + bv;
        if (OUT_F16)
          ((_Float16*)Cout)[(size_t)(mrow + r) * ldc + n] = (_Float16)v;
        else
          ((float*)Cout)[(size_t)(mrow + r) * ldc + n] = v;
      }
    }
  }
}

// ---------------------------------------------------------------------------
// LayerNorm over D=64 on q and k slices of qkv (in place, f16 storage,
// f32 math). One wave per (b,n,{q|k},h) row. q is pre-scaled by D^-0.5.
// ---------------------------------------------------------------------------
__global__ void ln_qk_kernel(_Float16* __restrict__ qkv,
                             const float* __restrict__ qg, const float* __restrict__ qb,
                             const float* __restrict__ kg, const float* __restrict__ kb) {
  const int wid  = (blockIdx.x * blockDim.x + threadIdx.x) >> 5;
  const int lane = threadIdx.x & 31;
  // wid in [0, B*N*2*H): decompose
  const int bn  = wid >> 5;        // / (2*H)
  const int rem = wid & 31;
  const int qk  = rem >> 4;        // 0 = q, 1 = k
  const int h   = rem & 15;
  _Float16* p = qkv + (size_t)bn * LD3 + qk * C_ + h * D_;

  const int d0 = lane * 2;
  float x0 = (float)p[d0], x1 = (float)p[d0 + 1];
  float s = x0 + x1, sq = x0 * x0 + x1 * x1;
#pragma unroll
  for (int off = 16; off; off >>= 1) {
    s  += __shfl_xor(s,  off, 32);
    sq += __shfl_xor(sq, off, 32);
  }
  const float mean = s * (1.0f / 64.0f);
  const float var  = sq * (1.0f / 64.0f) - mean * mean;
  const float inv  = rsqrtf(var + 1e-5f);
  const float scale = qk ? 1.0f : 0.125f;   // D^-0.5 = 1/8 applied to q
  const float* g  = qk ? kg : qg;
  const float* bb = qk ? kb : qb;
  p[d0]     = (_Float16)(((x0 - mean) * inv * g[d0]     + bb[d0])     * scale);
  p[d0 + 1] = (_Float16)(((x1 - mean) * inv * g[d0 + 1] + bb[d0 + 1]) * scale);
}

// ---------------------------------------------------------------------------
// Fused attention per (b,h): S = q k^T -> softmax (write f32 weights) ->
// out = P v (write f16 atth). Block = 128 threads (4 waves) owns 16 rows.
// grid = (N/16 row blocks, B*H).
// Dynamic LDS layout (bytes):
//   [0,      66048)  S     f32 [16][1032]
//   [66048,  99072)  P     f16 [16][1032]
//   [99072, 104192)  vT    f16 [64][40]   (32 K values + pad)
//   [104192,104256)  invs  f32 [16]
// ---------------------------------------------------------------------------
#define SLD 1032
#define PLD 1032
#define VLD 40

__global__ void attn_kernel(const _Float16* __restrict__ qkv,
                            float* __restrict__ weights,      // [B,H,N,N]
                            _Float16* __restrict__ atth) {    // [B,N,C]
  extern __shared__ char smem[];
  float*    Slds = (float*)smem;
  _Float16* Plds = (_Float16*)(smem + 66048);
  _Float16* vT   = (_Float16*)(smem + 99072);
  float*    invs = (float*)(smem + 104192);

  const int lane = threadIdx.x & 31;
  const int wave = threadIdx.x >> 5;          // 0..3
  const int n0   = blockIdx.x;                // row block (16 rows)
  const int b    = blockIdx.y >> 4;
  const int h    = blockIdx.y & 15;

  const _Float16* qbase = qkv + ((size_t)b * N_ + n0 * 16) * LD3 + h * D_;
  const _Float16* kbase = qkv + (size_t)b * N_ * LD3 + C_ + h * D_;
  const _Float16* vbase = qkv + (size_t)b * N_ * LD3 + 2 * C_ + h * D_;

  // ---- Phase 1: S tile strip, 16 x 1024 (each wave: 16 column tiles) ----
  for (int t = 0; t < 16; ++t) {
    const int ct = wave * 16 + t;
    v8f acc = {};
#pragma unroll
    for (int k0 = 0; k0 < D_; k0 += 32) {
      v16h a = load_frag_K(qbase, LD3, 0,       k0, lane);  // q rows
      v16h bfr = load_frag_K(kbase, LD3, ct * 16, k0, lane); // k rows = B cols
      acc = __builtin_amdgcn_wmma_f32_16x16x32_f16(false, a, false, bfr,
                                                   (short)0, acc, false, false);
    }
    const int nn = ct * 16 + (lane & 15);
    const int mb = (lane >> 4) << 3;
#pragma unroll
    for (int r = 0; r < 8; ++r) Slds[(mb + r) * SLD + nn] = acc[r];
  }
  __syncthreads();

  // ---- Phase 2a: rowwise max & expsum (8 threads per row) ----
  {
    const int r   = threadIdx.x >> 3;    // 0..15
    const int sub = threadIdx.x & 7;
    float mx = -3.0e38f;
    for (int c = sub; c < N_; c += 8) mx = fmaxf(mx, Slds[r * SLD + c]);
#pragma unroll
    for (int off = 4; off; off >>= 1) mx = fmaxf(mx, __shfl_xor(mx, off, 32));
    float sum = 0.0f;
    for (int c = sub; c < N_; c += 8) {
      float e = __expf(Slds[r * SLD + c] - mx);
      Slds[r * SLD + c] = e;             // overwrite score with exp
      sum += e;
    }
#pragma unroll
    for (int off = 4; off; off >>= 1) sum += __shfl_xor(sum, off, 32);
    if (sub == 0) invs[r] = 1.0f / sum;
  }
  __syncthreads();

  // ---- Phase 2b: coalesced weight store + f16 probs to LDS ----
  {
    float* wout = weights + ((size_t)blockIdx.y * N_ + n0 * 16) * N_;
    for (int idx = threadIdx.x; idx < 16 * N_; idx += 128) {
      const int r = idx >> 10, c = idx & (N_ - 1);
      const float p = Slds[r * SLD + c] * invs[r];
      wout[(size_t)r * N_ + c] = p;
      Plds[r * PLD + c] = (_Float16)p;
    }
  }

  // ---- Phase 3: out = P @ v, each wave owns one 16-wide d tile ----
  v8f acc = {};
  for (int k0 = 0; k0 < N_; k0 += 32) {
    // stage v[k0..k0+31][0..63] transposed into vT[d][kk]
#pragma unroll
    for (int i = 0; i < 16; ++i) {
      const int linear = threadIdx.x * 16 + i;    // 2048 elements
      const int kk = linear >> 6, d = linear & 63;
      vT[d * VLD + kk] = vbase[(size_t)(k0 + kk) * LD3 + d];
    }
    __syncthreads();
    v16h a = load_frag_K(Plds, PLD, 0, k0, lane);
    v16h bfr = load_frag_K(vT, VLD, wave * 16, 0, lane);
    acc = __builtin_amdgcn_wmma_f32_16x16x32_f16(false, a, false, bfr,
                                                 (short)0, acc, false, false);
    __syncthreads();
  }
  {
    const int col = h * D_ + wave * 16 + (lane & 15);
    const int mb = (lane >> 4) << 3;
#pragma unroll
    for (int r = 0; r < 8; ++r) {
      const size_t row = (size_t)b * N_ + n0 * 16 + mb + r;
      atth[row * C_ + col] = (_Float16)acc[r];
    }
  }
}

// ---------------------------------------------------------------------------
// Host-side launch
// ---------------------------------------------------------------------------
extern "C" void kernel_launch(void* const* d_in, const int* in_sizes, int n_in,
                              void* d_out, int out_size, void* d_ws, size_t ws_size,
                              hipStream_t stream) {
  const float* x      = (const float*)d_in[0];
  const float* W_qkv  = (const float*)d_in[1];
  const float* b_qkv  = (const float*)d_in[2];
  const float* qn_g   = (const float*)d_in[3];
  const float* qn_b   = (const float*)d_in[4];
  const float* kn_g   = (const float*)d_in[5];
  const float* kn_b   = (const float*)d_in[6];
  const float* W_proj = (const float*)d_in[7];
  const float* b_proj = (const float*)d_in[8];

  // workspace layout (needs 48 MB)
  char* ws = (char*)d_ws;
  _Float16* xh    = (_Float16*)(ws + 0);                        // 4M f16 (8MB)
  _Float16* wqkvh = (_Float16*)(ws + (size_t)8  * 1024 * 1024); // 3M f16 (6MB)
  _Float16* wprjh = (_Float16*)(ws + (size_t)14 * 1024 * 1024); // 1M f16 (2MB)
  _Float16* qkvh  = (_Float16*)(ws + (size_t)16 * 1024 * 1024); // 12M f16 (24MB)
  _Float16* atth  = (_Float16*)(ws + (size_t)40 * 1024 * 1024); // 4M f16 (8MB)

  float* out_proj = (float*)d_out;                              // [B,N,C]
  float* out_w    = out_proj + (size_t)B_ * N_ * C_;            // [B,H,N,N]

  const int n_x = B_ * N_ * C_;          // 4194304
  const int n_wq = 3 * C_ * C_;          // 3145728
  const int n_wp = C_ * C_;              // 1048576

  f32_to_f16_kernel<<<(n_x  + 255) / 256, 256, 0, stream>>>(x,      xh,    n_x);
  f32_to_f16_kernel<<<(n_wq + 255) / 256, 256, 0, stream>>>(W_qkv,  wqkvh, n_wq);
  f32_to_f16_kernel<<<(n_wp + 255) / 256, 256, 0, stream>>>(W_proj, wprjh, n_wp);

  // QKV GEMM: [4096 x 1024] @ [3072 x 1024]^T -> f16 [4096 x 3072] (+bias)
  // 32x64 per wave: grid.x = 3072/64/8 = 6, grid.y = 4096/32 = 128
  wmma_gemm_kernel<true><<<dim3(6, 128), 256, 0, stream>>>(
      xh, C_, wqkvh, C_, b_qkv, qkvh, LD3, C_);

  // LayerNorm q,k: 131072 wave-rows, 8 waves per block
  ln_qk_kernel<<<16384, 256, 0, stream>>>(qkvh, qn_g, qn_b, kn_g, kn_b);

  // Fused attention: 64 row-blocks x (B*H = 64)
  attn_kernel<<<dim3(N_ / 16, B_ * H_), 128, 104256, stream>>>(
      qkvh, out_w, atth);

  // Projection GEMM: [4096 x 1024] @ [1024 x 1024]^T -> f32 d_out (+bias)
  // grid.x = 1024/64/8 = 2, grid.y = 4096/32 = 128
  wmma_gemm_kernel<false><<<dim3(2, 128), 256, 0, stream>>>(
      atth, C_, wprjh, C_, b_proj, out_proj, C_, C_);
}
